// HpNSynth_16277926051937
// MI455X (gfx1250) — matrix operand order repaired and checked by reading, not compiled, not run on hardware
//
#include <hip/hip_runtime.h>
#include <hip/hip_bf16.h>
#include <math.h>

#define SR     16000
#define NBATCH 8
#define TFR    1000
#define NH     100
#define NBANDS 33
#define BLK    64
#define NSAMP  64000      // TFR*BLK
#define RLEN   16000
#define LOG10F 2.302585092994046f
#define PIF    3.14159265358979323846f

// reverb WMMA geometry (time-reversed signal, padded filter)
#define XP_PITCH 80032    // per-batch reversed+padded f16 signal (mult of 16)
#define XOFF     64015    // xr[XOFF - n] = x[n]
#define HP_PAD   256      // front zero pad of filter (covers c0 = -256)
#define HP_LEN   16512    // 256 + 16000 taps + tail zeros
#define C0_MIN   (-256)
#define C0_MAX   15968

typedef _Float16 half_t;
typedef __attribute__((ext_vector_type(16))) _Float16     v16h;
typedef __attribute__((ext_vector_type(8)))  float        v8f;
typedef __attribute__((ext_vector_type(4)))  unsigned int v4u;
typedef __attribute__((ext_vector_type(8)))  int          v8i;
typedef __attribute__((ext_vector_type(4)))  int          v4i;

__device__ __forceinline__ float exp_sigmoid(float x) {
    float s = 1.0f / (1.0f + __expf(-x));
    return 2.0f * __powf(s, LOG10F) + 1e-7f;
}

// ---------------- K0: zero the f16 scratch (xr + hp are contiguous) -------
__global__ void zero_kernel(unsigned int* p, int n) {
    int i = blockIdx.x * 256 + threadIdx.x;
    if (i < n) p[i] = 0u;
}

// ---------------- K1: controls + per-block noise impulse ------------------
__global__ void __launch_bounds__(128) controls_kernel(
        const float* __restrict__ hd, const float* __restrict__ nb,
        const float* __restrict__ amp, const float* __restrict__ f0,
        float* __restrict__ out_harm, float* __restrict__ out_nb,
        float* __restrict__ out_f0, float* __restrict__ nimp) {
    int bt  = blockIdx.x;          // b*TFR + t
    int tid = threadIdx.x;         // 128
    __shared__ float red[128];
    __shared__ float nbs[NBANDS];
    __shared__ float s_amp, s_f0;
    if (tid == 0) {
        s_amp = exp_sigmoid(amp[bt]);
        s_f0  = f0[bt];
        out_f0[bt] = f0[bt];
    }
    __syncthreads();
    float hv = 0.0f;
    if (tid < NH) {
        float aa = ((s_f0 * (float)(tid + 1) < (float)SR * 0.5f) ? 1.0f : 0.0f) + 1e-4f;
        hv = exp_sigmoid(hd[bt * NH + tid]) * aa;
    }
    red[tid] = hv;
    if (tid < NBANDS) {
        float v = exp_sigmoid(nb[bt * NBANDS + tid]);
        nbs[tid] = v;
        out_nb[bt * NBANDS + tid] = v;
    }
    __syncthreads();
    #pragma unroll
    for (int off = 64; off > 0; off >>= 1) {
        if (tid < off) red[tid] += red[tid + off];
        __syncthreads();
    }
    float tot = red[0];
    if (tid < NH) out_harm[bt * NH + tid] = hv / tot * s_amp;
    __syncthreads();
    // irfft(33 real bins) * rolled Hann -> 64-tap impulse
    if (tid < BLK) {
        int n = tid;
        float acc = nbs[0];
        #pragma unroll 8
        for (int k = 1; k < 32; ++k)
            acc += 2.0f * nbs[k] * __cosf((2.0f * PIF / 64.0f) * (float)k * (float)n);
        acc += nbs[32] * ((n & 1) ? -1.0f : 1.0f);
        acc *= (1.0f / 64.0f);
        float win = 0.5f + 0.5f * __cosf(PIF * (float)n / 32.0f);  // win[(n+32)%64]
        nimp[bt * BLK + n] = acc * win;
    }
}

// ---------------- K2: f0 upsample + phase cumsum (per-batch scan) ---------
__global__ void __launch_bounds__(256) phase_kernel(const float* __restrict__ f0,
                                                    float* __restrict__ omega) {
    int b   = blockIdx.x;
    int tid = threadIdx.x;   // 256
    __shared__ float buf[256];
    __shared__ float carry;
    if (tid == 0) carry = 0.0f;
    __syncthreads();
    for (int base = 0; base < NSAMP; base += 256) {
        int n = base + tid;
        float pos = ((float)n + 0.5f) * (1.0f / 64.0f) - 0.5f;
        pos = fminf(fmaxf(pos, 0.0f), (float)(TFR - 1));
        int   i0 = (int)pos;
        int   i1 = min(i0 + 1, TFR - 1);
        float w  = pos - (float)i0;
        float f  = f0[b * TFR + i0] * (1.0f - w) + f0[b * TFR + i1] * w;
        buf[tid] = 2.0f * PIF * f / (float)SR;
        __syncthreads();
        #pragma unroll
        for (int off = 1; off < 256; off <<= 1) {
            float t = (tid >= off) ? buf[tid - off] : 0.0f;
            __syncthreads();
            buf[tid] += t;
            __syncthreads();
        }
        float c = carry;
        omega[b * NSAMP + n] = c + buf[tid];
        __syncthreads();
        if (tid == 255) carry = c + buf[255];
        __syncthreads();
    }
}

// ---------------- K2b: reverb impulse taps (f16, identity tap zeroed) -----
__global__ void hfill_kernel(const float* __restrict__ rnoise,
                             const float* __restrict__ decay,
                             const float* __restrict__ wet,
                             half_t* __restrict__ hp) {
    int j = blockIdx.x * 256 + threadIdx.x;
    if (j >= RLEN) return;
    if (j == 0) { hp[HP_PAD] = (half_t)0.0f; return; }  // identity tap added in f32
    float dk = -log1pf(__expf(-decay[0])) * 500.0f / (float)SR;
    float wg = 1.0f / (1.0f + __expf(-wet[0]));
    hp[HP_PAD + j] = (half_t)(rnoise[j] * __expf(dk * (float)j) * wg);
}

// ---------------- K3: harmonic oscillator bank ---------------------------
__global__ void __launch_bounds__(64) harmonic_kernel(const float* __restrict__ omega,
                                                      const float* __restrict__ harm,
                                                      float* __restrict__ out_hsig) {
    int bt = blockIdx.x;
    int b  = bt / TFR, t = bt % TFR;
    int tid = threadIdx.x;  // 64
    __shared__ float as[NH];
    for (int i = tid; i < NH; i += 64) as[i] = harm[bt * NH + i];
    __syncthreads();
    float om  = omega[b * NSAMP + t * BLK + tid];
    float acc = 0.0f;
    #pragma unroll 4
    for (int h = 0; h < NH; ++h)
        acc += __sinf(om * (float)(h + 1)) * as[h];
    out_hsig[b * NSAMP + t * BLK + tid] = acc;
}

// ---------------- K4: per-block noise FIR + dereverb + reversed f16 -------
__global__ void __launch_bounds__(64) noise_kernel(const float* __restrict__ noise,
                                                   const float* __restrict__ nimp,
                                                   const float* __restrict__ hsig,
                                                   float* __restrict__ out_nsig,
                                                   float* __restrict__ out_der,
                                                   half_t* __restrict__ xr) {
    int bt = blockIdx.x;
    int b  = bt / TFR, t = bt % TFR;
    int m  = threadIdx.x;   // 64
    __shared__ float si[BLK], ss[BLK];
    si[m] = nimp[bt * BLK + m];
    ss[m] = noise[bt * BLK + m] * 2.0f - 1.0f;
    __syncthreads();
    float acc = 0.0f;
    for (int k = 0; k <= m; ++k) acc += si[k] * ss[m - k];
    int n = t * BLK + m;
    out_nsig[b * NSAMP + n] = acc;
    float d = acc + hsig[b * NSAMP + n];
    out_der[b * NSAMP + n] = d;
    xr[b * XP_PITCH + (XOFF - n)] = (half_t)d;   // time-reversed f16 stage
}

// ---------------- K5: 16000-tap reverb FIR via WMMA f16 ------------------
// A = reversed signal (global, contiguous 16B runs per lane),
// B = filter window  (LDS, 32B-aligned contiguous run per lane),
// D[m][col] = out_b[n0 + 16*col + m]. One wave = 256 contiguous outputs.
__global__ void __launch_bounds__(128) reverb_wmma_kernel(
        const half_t* __restrict__ xr, const half_t* __restrict__ hp,
        const float* __restrict__ der, float* __restrict__ out) {
    __shared__ __align__(16) half_t hlds[HP_LEN];
    int tid  = threadIdx.x;
    int wid  = tid >> 5;
    int lane = tid & 31;

#if __has_builtin(__builtin_amdgcn_tensor_load_to_lds) && __has_builtin(__builtin_amdgcn_s_wait_tensorcnt)
    if (wid == 0) {   // TDM: pull the whole filter (33 KB) into LDS once/block
        unsigned long long ga = (unsigned long long)(uintptr_t)hp;
        v4u g0;
        g0[0] = 1u;                                       // count=1, user mode
        g0[1] = (unsigned int)(uintptr_t)hlds;            // lds_addr
        g0[2] = (unsigned int)(ga & 0xffffffffu);         // global_addr[31:0]
        g0[3] = (unsigned int)((ga >> 32) & 0x01ffffffu)  // global_addr[56:32]
              | (2u << 30);                               // type = 2 (image)
        v8i g1;
        g1[0] = (1 << 16);                       // data_size = 1 (2B), no mcast
        g1[1] = (int)((HP_LEN & 0xffff) << 16);  // tensor_dim0[15:0]
        g1[2] = (1 << 16);                       // tensor_dim1 = 1
        g1[3] = (int)(HP_LEN << 16);             // tile_dim0 = HP_LEN
        g1[4] = 1;                               // tile_dim1 = 1
        g1[5] = HP_LEN;                          // tensor_dim0_stride
        g1[6] = 0;
        g1[7] = 0;
        v4i gz4 = {0, 0, 0, 0};
        v8i gz8 = {0, 0, 0, 0, 0, 0, 0, 0};
        __builtin_amdgcn_tensor_load_to_lds(g0, g1, gz4, gz4, gz8, 0);
        __builtin_amdgcn_s_wait_tensorcnt(0);
    }
#else
    for (int i = tid; i < HP_LEN; i += 128) hlds[i] = hp[i];
#endif
    __syncthreads();

    int task = blockIdx.x * 4 + wid;   // 2000 wave-tasks
    int b    = task / 250;
    int n0   = (task % 250) * 256;

    int M    = lane & 15;              // A row / B column / D column index
    int hi16 = lane >> 4;

    // A: two contiguous 16B runs per lane from reversed signal
    //    run base = XOFF - n0 - M + c0 + hi16*16  (+8 for second run)
    const half_t* xbase = xr + (size_t)b * XP_PITCH
                             + (size_t)(XOFF - n0 - M + hi16 * 16);
    // B: one contiguous 32B run per lane from LDS filter
    //    base = HP_PAD + c0 + hi16*16 + 16*M
    const half_t* hbase = hlds + (HP_PAD + hi16 * 16 + 16 * M);

    v8f acc = {};
    for (int c0 = C0_MIN; c0 <= C0_MAX; c0 += 32) {
        v16h A, Bv;
        const half_t* xa = xbase + c0;
        const half_t* hb = hbase + c0;
        #pragma unroll
        for (int e = 0; e < 8; ++e) {
            A[e]     = xa[e];          // run 1: K = hi16*16 + e
            A[8 + e] = xa[8 + e];      // run 2: K = hi16*16 + 8 + e
        }
        #pragma unroll
        for (int e = 0; e < 16; ++e)
            Bv[e] = hb[e];             // K = hi16*16 + e
        acc = __builtin_amdgcn_wmma_f32_16x16x32_f16(
                  false, A, false, Bv, (short)0, acc, false, false);
    }

    // D: VGPR r -> m = r + 8*hi16, col = lane&15; outputs are contiguous.
    // Add the f32 identity tap (imp[0]=1): synth = dereverb + f16 tail.
    int nb0 = n0 + 16 * M + 8 * hi16;
    #pragma unroll
    for (int r = 0; r < 8; ++r) {
        int n = nb0 + r;
        out[b * NSAMP + n] = acc[r] + der[b * NSAMP + n];
    }
}

extern "C" void kernel_launch(void* const* d_in, const int* in_sizes, int n_in,
                              void* d_out, int out_size, void* d_ws, size_t ws_size,
                              hipStream_t stream) {
    (void)in_sizes; (void)n_in; (void)out_size; (void)ws_size;
    const float* hd     = (const float*)d_in[0];   // 8x1000x100
    const float* nb     = (const float*)d_in[1];   // 8x1000x33
    const float* amp    = (const float*)d_in[2];   // 8x1000x1
    const float* f0     = (const float*)d_in[3];   // 8x1000x1
    const float* noise  = (const float*)d_in[4];   // 8x1000x64
    const float* rnoise = (const float*)d_in[5];   // 16000
    const float* decay  = (const float*)d_in[6];   // scalar
    const float* wet    = (const float*)d_in[7];   // scalar

    // outputs, concatenated in return order
    float* o_synth = (float*)d_out;                  // 8x64000
    float* o_der   = o_synth + NBATCH * NSAMP;       // 8x64000
    float* o_nsig  = o_der   + NBATCH * NSAMP;       // 8x64000
    float* o_hsig  = o_nsig  + NBATCH * NSAMP;       // 8x64000
    float* o_harm  = o_hsig  + NBATCH * NSAMP;       // 8x1000x100
    float* o_nb    = o_harm  + NBATCH * TFR * NH;    // 8x1000x33
    float* o_f0    = o_nb    + NBATCH * TFR * NBANDS;// 8x1000

    // workspace
    float*  omega = (float*)d_ws;                    // 512000 f32
    float*  nimp  = omega + NBATCH * NSAMP;          // 512000 f32
    half_t* xr    = (half_t*)(nimp + NBATCH * TFR * BLK); // 8*80032 f16 (reversed)
    half_t* hp    = xr + NBATCH * XP_PITCH;          // 16512 f16 (contiguous)

    int zero_words = (NBATCH * XP_PITCH + HP_LEN) / 2;
    zero_kernel<<<(zero_words + 255) / 256, 256, 0, stream>>>((unsigned int*)xr, zero_words);

    controls_kernel<<<NBATCH * TFR, 128, 0, stream>>>(hd, nb, amp, f0,
                                                      o_harm, o_nb, o_f0, nimp);
    hfill_kernel<<<(RLEN + 255) / 256, 256, 0, stream>>>(rnoise, decay, wet, hp);
    phase_kernel<<<NBATCH, 256, 0, stream>>>(f0, omega);
    harmonic_kernel<<<NBATCH * TFR, 64, 0, stream>>>(omega, o_harm, o_hsig);
    noise_kernel<<<NBATCH * TFR, 64, 0, stream>>>(noise, nimp, o_hsig,
                                                  o_nsig, o_der, xr);
    reverb_wmma_kernel<<<500, 128, 0, stream>>>(xr, hp, o_der, o_synth);
}
